// Net_60198261621214
// MI455X (gfx1250) — matrix operand order repaired and checked by reading, not compiled, not run on hardware
//
#include <hip/hip_runtime.h>
#include <hip/hip_bf16.h>
#include <cstdint>
#include <cstddef>

// ---------------------------------------------------------------------------
// MI455X (gfx1250) implementation.
//  * All conv layers -> implicit GEMM on v_wmma_f32_16x16x32_f16 (f16 in, f32 acc)
//  * Weights pre-packed per launch into [CoutPad64][KtotPad32] f16 panels
//  * Fast path: 2-wave block; A tile LDS-staged cooperatively; B tile staged by
//    the Tensor Data Mover (tensor_load_to_lds, double-buffered, s_wait_tensorcnt)
//  * Activations kept in f16 end-to-end (working set < 100MB => L2-resident)
//  * PAM attention: softmax-invariant mean-subtraction dropped; 64-thr/pixel
// Workspace assumption: ws_size >= ~96MB (weights ~4MB + activations ~70MB).
// ---------------------------------------------------------------------------

typedef __attribute__((ext_vector_type(16))) _Float16 v16h;
typedef __attribute__((ext_vector_type(8)))  _Float16 v8h;
typedef __attribute__((ext_vector_type(8)))  float    v8f;
typedef __attribute__((ext_vector_type(4)))  unsigned int v4u;
typedef __attribute__((ext_vector_type(8)))  int v8i;
typedef __attribute__((ext_vector_type(4)))  int v4i;

// ---------------- TDM helpers ----------------------------------------------
// 2D tile load: 64 rows x 32 halves, row stride Kpad halves, into LDS (compacted).
// D# fields per CDNA5 ISA ch.8: group0 {count, lds_addr, global_addr, type=2},
// group1 {data_size=1(2B), tensor_dim0=Kpad, tensor_dim1=64, tile_dim0=32,
//         tile_dim1=64, tensor_dim0_stride=Kpad}.
__device__ __forceinline__ void tdm_load_b_tile(const _Float16* gaddr, unsigned ldsoff, int Kpad) {
#if defined(__HIP_DEVICE_COMPILE__)
  unsigned long long ga = (unsigned long long)(uintptr_t)gaddr;
  v4u g0 = { 1u,                                   // count=1, gather off
             ldsoff,                               // lds_addr (bytes)
             (unsigned)ga,                         // global_addr[31:0]
             (unsigned)(ga >> 32) | 0x80000000u }; // global_addr[56:32] | type=2
  v8i g1 = { (int)0x00010000u,                           // data_size=1 (2 bytes)
             (int)(((unsigned)Kpad & 0xffffu) << 16),    // tensor_dim0[15:0]
             (int)((((unsigned)Kpad) >> 16) | (64u << 16)), // dim0 hi | tensor_dim1=64
             (int)(32u << 16),                           // tile_dim0=32
             64,                                         // tile_dim1=64, tile_dim2=0
             Kpad,                                       // tensor_dim0_stride lo
             0, 0 };
  v4i z4 = { 0, 0, 0, 0 };
#if __clang_major__ >= 23
  v8i z8 = { 0, 0, 0, 0, 0, 0, 0, 0 };
  __builtin_amdgcn_tensor_load_to_lds(g0, g1, z4, z4, z8, 0);
#else
  __builtin_amdgcn_tensor_load_to_lds(g0, g1, z4, z4, 0);
#endif
#else
  (void)gaddr; (void)ldsoff; (void)Kpad;
#endif
}

__device__ __forceinline__ void wait_tensor0() {
#if defined(__HIP_DEVICE_COMPILE__)
  __builtin_amdgcn_s_wait_tensorcnt(0);
#endif
}

// ---------------- elementwise / packing kernels ----------------------------

__global__ void k_pack_w(const float* __restrict__ w, _Float16* __restrict__ dst,
                         int Cout, int Ktot, int Kpad, int CoutPad) {
  int i = blockIdx.x * 256 + threadIdx.x;
  int total = CoutPad * Kpad;
  if (i >= total) return;
  int co = i / Kpad, k = i - co * Kpad;
  float v = (co < Cout && k < Ktot) ? w[(size_t)co * Ktot + k] : 0.f;
  dst[i] = (_Float16)v;
}

__global__ void k_cvt_f32_f16(const float* __restrict__ in, _Float16* __restrict__ out, int n) {
  int i = blockIdx.x * 256 + threadIdx.x;
  if (i < n) out[i] = (_Float16)in[i];
}

__global__ void k_copy_f16(const _Float16* __restrict__ in, _Float16* __restrict__ out, int n) {
  int i = blockIdx.x * 256 + threadIdx.x;
  if (i < n) out[i] = in[i];
}

// ---------------- im2col helper --------------------------------------------
template <int KK>
__device__ __forceinline__ _Float16
im2col_at(const _Float16* __restrict__ in, size_t cibase, int k, int Ktot,
          int hh, int ww, int sp, int H, int W, int HWt, int pad, bool mvalid) {
  constexpr int Ksz = (KK == 9) ? 3 : 1;
  _Float16 v = (_Float16)0.f;
  if (mvalid && k < Ktot) {
    if (KK == 1) {
      v = in[(cibase + k) * HWt + sp];
    } else {
      int ci = k / KK, r = k - ci * KK;          // KK compile-time: cheap
      int kh = r / Ksz, kw = r - kh * Ksz;
      int ih = hh + kh - pad, iw = ww + kw - pad;
      if (ih >= 0 && ih < H && iw >= 0 && iw < W)
        v = in[(cibase + ci) * HWt + (size_t)ih * W + iw];
    }
  }
  return v;
}

// ---------------- WMMA conv, fast path -------------------------------------
// 2 waves / block. Block computes 16(spatial) x 64(out-channel).
// A tile (16x32 f16) staged in LDS once per K-step, shared by 4 WMMAs.
// B tile (64x32 f16) staged by TDM, double-buffered.
template <int KK>
__global__ void __launch_bounds__(64)
k_conv_fast(const _Float16* __restrict__ in, const _Float16* __restrict__ wt,
            const float* __restrict__ bias, _Float16* __restrict__ out,
            const _Float16* __restrict__ resid,
            int H, int W, int Cin, int Cout, int pad, int groups, int act) {
  const int tid  = threadIdx.x;
  const int wv   = tid >> 5;
  const int lane = tid & 31;
  const int HWt  = H * W;
  const int M0   = blockIdx.x * 16;
  const int N0b  = blockIdx.y * 64;
  const int Cig  = Cin / groups;
  const int Cog  = Cout / groups;          // fast path requires 64 | Cog or g==1
  int g = N0b / Cog; if (g >= groups) g = groups - 1;
  const int Ktot = Cig * KK;
  const int Kpad = (Ktot + 31) & ~31;
  const size_t cibase = (size_t)g * Cig;

  __shared__ __align__(16) _Float16 As[16 * 32];
  __shared__ __align__(64) _Float16 Bs[2][64 * 32];

  // --- staging role: thread t stages row m=t>>2, k-run (t&3)*8 .. +7 ---
  const int  sm  = tid >> 2;
  const int  sk0 = (tid & 3) * 8;
  const int  ssp = M0 + sm;
  const int  shh = ssp / W;
  const int  sww = ssp - shh * W;
  const bool smv = (ssp < HWt);

  // --- compute role ---
  const int kAoff = (lane >> 4) * 16;
  const int nloc  = lane & 15;
  const int kBoff = (lane >> 4) * 16;
  const _Float16* wpanel = wt + (size_t)N0b * Kpad;     // block's 64-row panel
  const _Float16* arow   = &As[(size_t)(lane & 15) * 32 + kAoff];
  const int brow0 = (wv * 32 + nloc) * 32 + kBoff;      // LDS half-index, tile 0
  const int brow1 = brow0 + 16 * 32;                    // tile 1 (+16 rows)

  // prologue: wave0 kicks off the first B tile
  if (wv == 0) tdm_load_b_tile(wpanel, (unsigned)(uintptr_t)&Bs[0][0], Kpad);

  v8f acc0 = {}, acc1 = {};
  int cur = 0;
  for (int kb = 0; kb < Kpad; kb += 32, cur ^= 1) {
    v8h sv;
#pragma unroll
    for (int u = 0; u < 8; ++u)
      sv[u] = im2col_at<KK>(in, cibase, kb + sk0 + u, Ktot, shh, sww, ssp, H, W, HWt, pad, smv);
    *(v8h*)&As[(size_t)sm * 32 + sk0] = sv;              // ds_store_b128
    wait_tensor0();                                      // s_wait_tensorcnt 0
    __syncthreads();                                     // A + B[cur] visible

    if (wv == 0 && kb + 32 < Kpad)                       // async-prefetch next B
      tdm_load_b_tile(wpanel + (size_t)(kb + 32), (unsigned)(uintptr_t)&Bs[cur ^ 1][0], Kpad);

    v8h lo = *(const v8h*)(arow);                        // ds_load_b128
    v8h hi = *(const v8h*)(arow + 8);                    // ds_load_b128
    v16h afrag;
#pragma unroll
    for (int u = 0; u < 8; ++u) { afrag[u] = lo[u]; afrag[u + 8] = hi[u]; }
    v16h b0 = *(const v16h*)&Bs[cur][brow0];             // ds_load_b128 x2
    v16h b1 = *(const v16h*)&Bs[cur][brow1];
    acc0 = __builtin_amdgcn_wmma_f32_16x16x32_f16(false, afrag, false, b0, (short)0, acc0, false, false);
    acc1 = __builtin_amdgcn_wmma_f32_16x16x32_f16(false, afrag, false, b1, (short)0, acc1, false, false);
    __syncthreads();                                     // reads done before overwrite
  }

  const int mo = (lane >> 4) * 8;
#pragma unroll
  for (int tt = 0; tt < 2; ++tt) {
    int co = N0b + wv * 32 + tt * 16 + nloc;
    if (co < Cout) {
      float bv = bias ? bias[co] : 0.f;
      v8f a = tt ? acc1 : acc0;
#pragma unroll
      for (int jj = 0; jj < 8; ++jj) {
        int sp2 = M0 + mo + jj;
        if (sp2 < HWt) {
          float v = a[jj] + bv;
          if (act == 1) v = (v >= 0.f) ? v : 0.1f * v;      // LeakyReLU(0.1)
          if (resid) v += (float)resid[(size_t)co * HWt + sp2];
          out[(size_t)co * HWt + sp2] = (_Float16)v;
        }
      }
    }
  }
}

// ---------------- WMMA conv, grouped fallback (Cog not 64-aligned) ---------
template <int KK>
__global__ void __launch_bounds__(32)
k_conv_g(const _Float16* __restrict__ in, const _Float16* __restrict__ wt,
         const float* __restrict__ bias, _Float16* __restrict__ out,
         const _Float16* __restrict__ resid,
         int H, int W, int Cin, int Cout, int pad, int groups, int act) {
  const int lane = threadIdx.x;
  const int HWt  = H * W;
  const int M0   = blockIdx.x * 16;
  const int N0   = blockIdx.y * 16;
  const int Cig  = Cin / groups;
  const int Cog  = Cout / groups;
  int g = N0 / Cog; if (g >= groups) g = groups - 1;
  const int Ktot = Cig * KK;
  const int Kpad = (Ktot + 31) & ~31;
  const size_t cibase = (size_t)g * Cig;

  const int  mloc = lane & 15;
  const int  sp   = M0 + mloc;
  const int  hh   = sp / W;
  const int  ww   = sp - hh * W;
  const bool mvalid = (sp < HWt);
  const int  kAoff = (lane >> 4) * 16;
  const int  nloc  = lane & 15;
  const int  kBoff = (lane >> 4) * 16;
  const _Float16* wrow = wt + (size_t)(N0 + nloc) * Kpad + kBoff;

  v8f acc = {};
  for (int kb = 0; kb < Kpad; kb += 32) {
    if (kb + 32 < Kpad) __builtin_prefetch(wrow + kb + 32, 0, 0);
    v16h bfrag = *(const v16h*)(wrow + kb);
    v16h afrag;
#pragma unroll
    for (int jj = 0; jj < 16; ++jj) {
      int k = kb + kAoff + jj;
      afrag[jj] = im2col_at<KK>(in, cibase, k, Ktot, hh, ww, sp, H, W, HWt, pad, mvalid);
    }
    acc = __builtin_amdgcn_wmma_f32_16x16x32_f16(false, afrag, false, bfrag, (short)0, acc, false, false);
  }

  const int co = N0 + nloc;
  if (co < Cout) {
    float bv = bias ? bias[co] : 0.f;
    const int mo = (lane >> 4) * 8;
#pragma unroll
    for (int jj = 0; jj < 8; ++jj) {
      int sp2 = M0 + mo + jj;
      if (sp2 < HWt) {
        float v = acc[jj] + bv;
        if (act == 1) v = (v >= 0.f) ? v : 0.1f * v;
        if (resid) v += (float)resid[(size_t)co * HWt + sp2];
        out[(size_t)co * HWt + sp2] = (_Float16)v;
      }
    }
  }
}

// ---------------- batchnorm (per channel over HW, B=1) ---------------------
__global__ void k_bn(const _Float16* __restrict__ x, const float* __restrict__ gamma,
                     const float* __restrict__ beta, _Float16* __restrict__ y, int HWt) {
  int c = blockIdx.x;
  __shared__ float s1[256], s2[256];
  float a = 0.f, b = 0.f;
  for (int i = threadIdx.x; i < HWt; i += 256) {
    float v = (float)x[(size_t)c * HWt + i];
    a += v; b += v * v;
  }
  s1[threadIdx.x] = a; s2[threadIdx.x] = b; __syncthreads();
  for (int st = 128; st > 0; st >>= 1) {
    if (threadIdx.x < st) { s1[threadIdx.x] += s1[threadIdx.x + st]; s2[threadIdx.x] += s2[threadIdx.x + st]; }
    __syncthreads();
  }
  float mu  = s1[0] / HWt;
  float var = s2[0] / HWt - mu * mu;
  float sc  = gamma[c] * rsqrtf(var + 1e-5f);
  float sh  = beta[c] - mu * sc;
  for (int i = threadIdx.x; i < HWt; i += 256) {
    float v = (float)x[(size_t)c * HWt + i];
    y[(size_t)c * HWt + i] = (_Float16)(v * sc + sh);
  }
}

// ---------------- PAM attention (8x8 window, 64 channels) ------------------
__global__ void k_pam(const _Float16* __restrict__ Q, const _Float16* __restrict__ Kf,
                      const _Float16* __restrict__ xl, const _Float16* __restrict__ xr,
                      const float* __restrict__ disp,
                      _Float16* __restrict__ outL, _Float16* __restrict__ outR,
                      int H, int W) {
  const int p = blockIdx.x;
  const int i = p / W, j = p - i * W;
  const int t = threadIdx.x;           // 0..63 : window slot, then channel
  const int HWt = H * W;
  __shared__ float q[64], kk[64], sc1[64], sc2[64], m1[64], m2[64];

  q[t]  = (float)Q [(size_t)t * HWt + p];
  kk[t] = (float)Kf[(size_t)t * HWt + p];
  int d = (int)disp[p];
  int jr = j - d; if (jr < 0) jr = 0;
  int jl = j + d; if (jl > W - 1) jl = W - 1;
  int Jr = jr + 4, Jl = jl + 4;
  bool rowok = (i + 4) < H;
  bool okr = rowok && (Jr < W);
  bool okl = rowok && (Jl < W);
  __syncthreads();

  int r = t >> 3, c2 = t & 7;
  int hr = i + r - 4, wr = Jr + c2 - 8;
  int wl = Jl + c2 - 8;
  bool vr = okr && hr >= 0 && hr < H && wr >= 0 && wr < W;
  bool vl = okl && hr >= 0 && hr < H && wl >= 0 && wl < W;
  int ir = vr ? hr * W + wr : 0;
  int il = vl ? hr * W + wl : 0;
  float s_r2l = 0.f, s_l2r = 0.f;
  for (int c = 0; c < 64; ++c) {
    float kv = vr ? (float)Kf[(size_t)c * HWt + ir] : 0.f;
    float qv = vl ? (float)Q [(size_t)c * HWt + il] : 0.f;
    s_r2l += q[c]  * kv;
    s_l2r += kk[c] * qv;
  }
  sc1[t] = s_r2l; sc2[t] = s_l2r; __syncthreads();

  if (t == 0) {
    float mx1 = -1e30f, mx2 = -1e30f;
    for (int n = 0; n < 64; ++n) { mx1 = fmaxf(mx1, sc1[n]); mx2 = fmaxf(mx2, sc2[n]); }
    float su1 = 0.f, su2 = 0.f;
    for (int n = 0; n < 64; ++n) {
      m1[n] = __expf(sc1[n] - mx1); su1 += m1[n];
      m2[n] = __expf(sc2[n] - mx2); su2 += m2[n];
    }
    float r1 = 1.f / su1, r2 = 1.f / su2;
    for (int n = 0; n < 64; ++n) { m1[n] *= r1; m2[n] *= r2; }
  }
  __syncthreads();

  float aL = 0.f, aR = 0.f;
  for (int n = 0; n < 64; ++n) {
    int rr = n >> 3, cc = n & 7;
    int h2 = i + rr - 4, w2 = Jr + cc - 8;
    bool v = okr && h2 >= 0 && h2 < H && w2 >= 0 && w2 < W;
    float xrv = v ? (float)xr[(size_t)t * HWt + h2 * W + w2] : 0.f;
    float xlv = v ? (float)xl[(size_t)t * HWt + h2 * W + w2] : 0.f;
    aL += m1[n] * xrv;
    aR += m2[n] * xlv;
  }
  float VL = (j - d >= 0)     ? 1.f : 0.f;
  float VR = (j + d <= W - 1) ? 1.f : 0.f;
  outL[(size_t)t * HWt + p] = (_Float16)((float)xl[(size_t)t * HWt + p] + aL * VL);
  outR[(size_t)t * HWt + p] = (_Float16)((float)xr[(size_t)t * HWt + p] + aR * VR);
}

// ---------------- channel attention helpers --------------------------------
__global__ void k_gap(const _Float16* __restrict__ x, float* __restrict__ y, int HWt) {
  int c = blockIdx.x;
  __shared__ float s[256];
  float a = 0.f;
  for (int i = threadIdx.x; i < HWt; i += 256) a += (float)x[(size_t)c * HWt + i];
  s[threadIdx.x] = a; __syncthreads();
  for (int st = 128; st > 0; st >>= 1) {
    if (threadIdx.x < st) s[threadIdx.x] += s[threadIdx.x + st];
    __syncthreads();
  }
  if (threadIdx.x == 0) y[c] = s[0] / HWt;
}

__global__ void k_ca_mlp(const float* __restrict__ y, const float* __restrict__ w1,
                         const float* __restrict__ b1, const float* __restrict__ w2,
                         const float* __restrict__ b2, float* __restrict__ out) {
  __shared__ float s1[8];
  int t = threadIdx.x;
  if (t < 8) {
    float a = b1[t];
    for (int i = 0; i < 128; ++i) a += w1[t * 128 + i] * y[i];
    s1[t] = (a >= 0.f) ? a : 0.1f * a;
  }
  __syncthreads();
  if (t < 128) {
    float a = b2[t];
    for (int j = 0; j < 8; ++j) a += w2[t * 8 + j] * s1[j];
    out[t] = 1.f / (1.f + __expf(-a));
  }
}

__global__ void k_scale_ch(_Float16* __restrict__ x, const float* __restrict__ y, int C, int HWt) {
  int i = blockIdx.x * 256 + threadIdx.x;
  if (i >= C * HWt) return;
  int c = i / HWt;
  x[i] = (_Float16)((float)x[i] * y[c]);
}

// ---------------- pixel shuffle r=4 ----------------------------------------
__global__ void k_shuffle(const _Float16* __restrict__ in, _Float16* __restrict__ out,
                          int H, int W) {
  int idx = blockIdx.x * 256 + threadIdx.x;
  int HWt = H * W;
  int total = 64 * 16 * HWt;
  if (idx >= total) return;
  int c   = idx / (16 * HWt);
  int r   = idx - c * 16 * HWt;
  int sub = r / HWt;
  int p   = r - sub * HWt;
  int r1 = sub >> 2, r2 = sub & 3;
  int h = p / W, w = p - h * W;
  out[(size_t)c * (HWt * 16) + (size_t)(h * 4 + r1) * (W * 4) + (w * 4 + r2)] =
      in[(size_t)(c * 16 + sub) * HWt + p];
}

// ---------------- bicubic x4 (Keys a=-0.5, half-pixel centers) -------------
__device__ __forceinline__ float cubw(float t) {
  float at = fabsf(t);
  if (at <= 1.f) return ((1.5f * at - 2.5f) * at) * at + 1.f;
  if (at < 2.f)  return (((-0.5f * at) + 2.5f) * at - 4.f) * at + 2.f;
  return 0.f;
}

__global__ void k_bicubic(const float* __restrict__ in, float* __restrict__ out, int H, int W) {
  int idx = blockIdx.x * 256 + threadIdx.x;
  int H4 = H * 4, W4 = W * 4;
  int total = 3 * H4 * W4;
  if (idx >= total) return;
  int c = idx / (H4 * W4);
  int p = idx - c * H4 * W4;
  int oy = p / W4, ox = p - oy * W4;
  float sy = (oy + 0.5f) * 0.25f - 0.5f;
  float sx = (ox + 0.5f) * 0.25f - 0.5f;
  int y0 = (int)floorf(sy), x0 = (int)floorf(sx);
  float acc = 0.f;
  for (int dy = -1; dy <= 2; ++dy) {
    int yy = y0 + dy; yy = yy < 0 ? 0 : (yy > H - 1 ? H - 1 : yy);
    float wy = cubw(sy - (float)(y0 + dy));
    for (int dx = -1; dx <= 2; ++dx) {
      int xx = x0 + dx; xx = xx < 0 ? 0 : (xx > W - 1 ? W - 1 : xx);
      float wx = cubw(sx - (float)(x0 + dx));
      acc += wy * wx * in[(size_t)c * H * W + (size_t)yy * W + xx];
    }
  }
  out[idx] = acc;
}

__global__ void k_final_add(const _Float16* __restrict__ c2, const float* __restrict__ bic,
                            float* __restrict__ out, int n) {
  int i = blockIdx.x * 256 + threadIdx.x;
  if (i < n) out[i] = (float)c2[i] + bic[i];
}

// ===========================================================================
//                                HOST SIDE
// ===========================================================================

struct CB { const float* w; const float* b; };
struct PackedW { const _Float16* w16; const float* bias; int Cout; int Ktot; };
struct RdbW { PackedW conv[4]; PackedW lff; };
struct RdgW { RdbW rdb[4]; PackedW conv; };

extern "C" void kernel_launch(void* const* d_in, const int* in_sizes, int n_in,
                              void* d_out, int out_size, void* d_ws, size_t ws_size,
                              hipStream_t stream) {
  (void)in_sizes; (void)n_in; (void)out_size; (void)ws_size;
  const int H = 64, W = 96, HWt = H * W;
  const int H4 = H * 4, W4 = W * 4, HW4 = H4 * W4;

  // ---- walk inputs in setup_inputs() insertion order ----
  int ip = 0;
  auto nf = [&]() -> const float* { return (const float*)d_in[ip++]; };
  auto ncb = [&]() -> CB { CB c; c.w = nf(); c.b = nf(); return c; };

  const float* xlF = nf();               // (1,4,64,96)
  const float* xrF = nf();
  CB p_init = ncb();                     // params['init']
  CB deep_c[4][4], deep_lff[4], deep_cv; // params['deep']
  for (int r = 0; r < 4; ++r) { for (int t = 0; t < 4; ++t) deep_c[r][t] = ncb(); deep_lff[r] = ncb(); }
  deep_cv = ncb();
  CB p_bq = ncb(), p_bs = ncb(), p_rb1 = ncb(), p_rb2 = ncb();   // params['pam']
  const float* bn_g = nf(); const float* bn_b = nf();
  CB fus_c[4], fus_lff;                                           // params['fusion']
  for (int t = 0; t < 4; ++t) fus_c[t] = ncb();
  fus_lff = ncb();
  const float* ca_w1 = nf(); const float* ca_b1 = nf();
  const float* ca_w2 = nf(); const float* ca_b2 = nf();
  CB fus_cv = ncb();
  CB rec_c[4][4], rec_lff[4], rec_cv;                             // params['rec']
  for (int r = 0; r < 4; ++r) { for (int t = 0; t < 4; ++t) rec_c[r][t] = ncb(); rec_lff[r] = ncb(); }
  rec_cv = ncb();
  CB up_c1 = ncb(), up_c2 = ncb();                                // params['up']

  // ---- workspace arena (bytes, 256-aligned) ----
  uintptr_t base = (uintptr_t)d_ws;
  size_t off = 0;
  auto ah = [&](size_t halves) -> _Float16* {
    _Float16* p = (_Float16*)(base + off);
    off += ((halves * 2 + 255) & ~(size_t)255);
    return p;
  };
  auto af = [&](size_t fl) -> float* {
    float* p = (float*)(base + off);
    off += ((fl * 4 + 255) & ~(size_t)255);
    return p;
  };

  // ---- weight packing (fp32 -> padded f16 GEMM panels, 64-row padded) ----
  auto pack = [&](CB cb, int Cout, int Ktot) -> PackedW {
    int Kpad = (Ktot + 31) & ~31, CP = (Cout + 63) & ~63;
    _Float16* dst = ah((size_t)CP * Kpad);
    int tot = CP * Kpad;
    k_pack_w<<<(tot + 255) / 256, 256, 0, stream>>>(cb.w, dst, Cout, Ktot, Kpad, CP);
    PackedW pw; pw.w16 = dst; pw.bias = cb.b; pw.Cout = Cout; pw.Ktot = Ktot;
    return pw;
  };

  PackedW w_init = pack(p_init, 64, 4 * 9);
  RdgW wdeep, wrec;
  for (int r = 0; r < 4; ++r) {
    for (int t = 0; t < 4; ++t) {
      wdeep.rdb[r].conv[t] = pack(deep_c[r][t], 24, (64 + t * 24) * 9);
      wrec.rdb[r].conv[t]  = pack(rec_c[r][t],  24, (64 + t * 24) * 9);
    }
    wdeep.rdb[r].lff = pack(deep_lff[r], 64, 160);
    wrec.rdb[r].lff  = pack(rec_lff[r],  64, 160);
  }
  wdeep.conv = pack(deep_cv, 64, 256);
  wrec.conv  = pack(rec_cv,  64, 256);
  PackedW w_rb1 = pack(p_rb1, 256, 64 * 9);      // grouped g=4: Cog=64 -> fast
  PackedW w_rb2 = pack(p_rb2, 256, 64 * 9);
  PackedW w_bq  = pack(p_bq, 64, 64);            // grouped g=4: Cog=16 -> fallback
  PackedW w_bs  = pack(p_bs, 64, 64);
  RdbW wfus;
  for (int t = 0; t < 4; ++t) wfus.conv[t] = pack(fus_c[t], 32, (128 + t * 32) * 9);
  wfus.lff = pack(fus_lff, 128, 256);
  PackedW w_fcv = pack(fus_cv, 64, 128);
  PackedW w_up1 = pack(up_c1, 1024, 64);
  PackedW w_up2 = pack(up_c2, 3, 64 * 9);

  // ---- activation buffers ----
  _Float16* x16[2]   = { ah(4 * (size_t)HWt),   ah(4 * (size_t)HWt)   };
  _Float16* feat[2]  = { ah(64 * (size_t)HWt),  ah(64 * (size_t)HWt)  };
  _Float16* dcat[2]  = { ah(256 * (size_t)HWt), ah(256 * (size_t)HWt) };
  _Float16* bfeat[2] = { ah(64 * (size_t)HWt),  ah(64 * (size_t)HWt)  };
  _Float16* fusin[2] = { ah(128 * (size_t)HWt), ah(128 * (size_t)HWt) };
  _Float16* scr256   = ah(256 * (size_t)HWt);   // rdb scratch (max 256 ch)
  _Float16* bnbuf    = ah(256 * (size_t)HWt);
  _Float16* t1       = ah(256 * (size_t)HWt);
  _Float16* t2       = ah(256 * (size_t)HWt);
  _Float16* Qb       = ah(64 * (size_t)HWt);
  _Float16* Kb       = ah(64 * (size_t)HWt);
  _Float16* fusmid   = ah(128 * (size_t)HWt);
  _Float16* flbuf    = ah(64 * (size_t)HWt);
  _Float16* reccat   = ah(256 * (size_t)HWt);
  _Float16* fl2      = ah(64 * (size_t)HWt);
  _Float16* up1      = ah(1024 * (size_t)HWt);
  _Float16* shuf     = ah(64 * (size_t)HWt * 16);
  _Float16* c2out    = ah(3 * (size_t)HW4);
  float*    gapy     = af(128);
  float*    cay      = af(128);
  float*    bic      = af(3 * (size_t)HW4);

  // ---- launch helpers ----
  auto conv = [&](const _Float16* in, const PackedW& pw, _Float16* out,
                  const _Float16* resid, int Hq, int Wq, int Cin, int Cout,
                  int Ksz, int pad, int groups, int act) {
    bool fast = (groups == 1) || ((Cout / groups) % 64 == 0);
    if (fast) {
      int CP = (Cout + 63) & ~63;
      dim3 grid((Hq * Wq + 15) / 16, CP / 64);
      if (Ksz == 3)
        k_conv_fast<9><<<grid, 64, 0, stream>>>(in, pw.w16, pw.bias, out, resid,
                                                Hq, Wq, Cin, Cout, pad, groups, act);
      else
        k_conv_fast<1><<<grid, 64, 0, stream>>>(in, pw.w16, pw.bias, out, resid,
                                                Hq, Wq, Cin, Cout, pad, groups, act);
    } else {
      int CP = (Cout + 15) & ~15;
      dim3 grid((Hq * Wq + 15) / 16, CP / 16);
      if (Ksz == 3)
        k_conv_g<9><<<grid, 32, 0, stream>>>(in, pw.w16, pw.bias, out, resid,
                                             Hq, Wq, Cin, Cout, pad, groups, act);
      else
        k_conv_g<1><<<grid, 32, 0, stream>>>(in, pw.w16, pw.bias, out, resid,
                                             Hq, Wq, Cin, Cout, pad, groups, act);
    }
  };
  auto copyh = [&](const _Float16* in, _Float16* out, size_t n) {
    k_copy_f16<<<((int)n + 255) / 256, 256, 0, stream>>>(in, out, (int)n);
  };
  auto run_rdb = [&](const _Float16* x, _Float16* outbuf, const RdbW& wr, int G0, int G) {
    copyh(x, scr256, (size_t)G0 * HWt);
    int ch = G0;
    for (int t = 0; t < 4; ++t) {
      conv(scr256, wr.conv[t], scr256 + (size_t)ch * HWt, nullptr, H, W, ch, G, 3, 1, 1, 1);
      ch += G;
    }
    conv(scr256, wr.lff, outbuf, x, H, W, ch, G0, 1, 0, 1, 0);
  };
  auto run_rdg = [&](const _Float16* x, _Float16* cat, _Float16* outfeat, const RdgW& wg) {
    const _Float16* cur = x;
    for (int r = 0; r < 4; ++r) {
      run_rdb(cur, cat + (size_t)r * 64 * HWt, wg.rdb[r], 64, 24);
      cur = cat + (size_t)r * 64 * HWt;
    }
    conv(cat, wg.conv, outfeat, nullptr, H, W, 256, 64, 1, 0, 1, 0);
  };

  // ---- stage 1: input convert, init conv, deep RDG (both eyes) ----
  const float* xin[2] = { xlF, xrF };
  for (int e = 0; e < 2; ++e) {
    k_cvt_f32_f16<<<(4 * HWt + 255) / 256, 256, 0, stream>>>(xin[e], x16[e], 4 * HWt);
    conv(x16[e], w_init, feat[e], nullptr, H, W, 4, 64, 3, 1, 1, 0);
    run_rdg(feat[e], dcat[e], bfeat[e], wdeep);
  }

  // ---- stage 2: PAM ----
  k_bn<<<256, 256, 0, stream>>>(dcat[0], bn_g, bn_b, bnbuf, HWt);
  conv(bnbuf, w_rb1, t1, nullptr, H, W, 256, 256, 3, 1, 4, 1);
  conv(t1,    w_rb2, t2, bnbuf,   H, W, 256, 256, 3, 1, 4, 0);
  conv(t2,    w_bq,  Qb, nullptr, H, W, 256, 64, 1, 0, 4, 0);
  k_bn<<<256, 256, 0, stream>>>(dcat[1], bn_g, bn_b, bnbuf, HWt);
  conv(bnbuf, w_rb1, t1, nullptr, H, W, 256, 256, 3, 1, 4, 1);
  conv(t1,    w_rb2, t2, bnbuf,   H, W, 256, 256, 3, 1, 4, 0);
  conv(t2,    w_bs,  Kb, nullptr, H, W, 256, 64, 1, 0, 4, 0);

  copyh(bfeat[0], fusin[0], (size_t)64 * HWt);
  copyh(bfeat[1], fusin[1], (size_t)64 * HWt);
  const float* disp = xlF + 3 * (size_t)HWt;     // x_left[:,3]
  k_pam<<<HWt, 64, 0, stream>>>(Qb, Kb, bfeat[0], bfeat[1], disp,
                                fusin[0] + (size_t)64 * HWt,
                                fusin[1] + (size_t)64 * HWt, H, W);

  // ---- stage 3: fusion + rec + upscale per eye ----
  for (int e = 0; e < 2; ++e) {
    run_rdb(fusin[e], fusmid, wfus, 128, 32);
    k_gap<<<128, 256, 0, stream>>>(fusmid, gapy, HWt);
    k_ca_mlp<<<1, 128, 0, stream>>>(gapy, ca_w1, ca_b1, ca_w2, ca_b2, cay);
    k_scale_ch<<<(128 * HWt + 255) / 256, 256, 0, stream>>>(fusmid, cay, 128, HWt);
    conv(fusmid, w_fcv, flbuf, nullptr, H, W, 128, 64, 1, 0, 1, 0);

    run_rdg(flbuf, reccat, fl2, wrec);

    conv(fl2, w_up1, up1, nullptr, H, W, 64, 1024, 1, 0, 1, 0);
    k_shuffle<<<(64 * 16 * HWt + 255) / 256, 256, 0, stream>>>(up1, shuf, H, W);
    conv(shuf, w_up2, c2out, nullptr, H4, W4, 64, 3, 3, 1, 1, 0);

    k_bicubic<<<(3 * HW4 + 255) / 256, 256, 0, stream>>>(xin[e], bic, H, W);
    k_final_add<<<(3 * HW4 + 255) / 256, 256, 0, stream>>>(
        c2out, bic, (float*)d_out + (size_t)e * 3 * HW4, 3 * HW4);
  }
}
